// AutoEncoder_31533649887292
// MI455X (gfx1250) — matrix operand order repaired and checked by reading, not compiled, not run on hardware
//
#include <hip/hip_runtime.h>
#include <hip/hip_fp16.h>

// ============================================================================
// AutoEncoder forward for MI455X (gfx1250, wave32).
//
// Roofline: batch=1; cost dominated by one-shot streaming of ~176MB fp32
// weights (512x43008 / 43008x512 linears) -> pure HBM bandwidth problem
// (~7.6us @ 23.3TB/s). Linears run on v_wmma_f32_16x16x32_f16: fp32 weights/
// activations converted to f16 in-register, f32 accumulate. K%64==0 for all
// layers -> 2 back-to-back WMMAs per loop iteration. Weight stream is
// read-once -> non-temporal loads (TH=NT) so it never evicts the reused
// activation set from the 192MB L2. Small-K linears stage the shared
// activation vector into LDS via gfx1250 async global->LDS DMA
// (global_load_async_to_lds_b128 + s_wait_asynccnt); the GEMV body is
// specialized per address space so LDS reads are ds_load_b128 and global
// reads are global_load_b128 (no flat / DScnt coupling in the stream loop).
// Small convs / BN / conv-transpose are latency-bound -> direct fp32 VALU.
//
// Input flattening (jax pytree: dict keys sorted, lists in order):
//   0..33   dec_params[0]   34..67 dec_params[1]
//     +0 b_in +1 b_out +2 ct1_b +3 ct1_w +4 ct2_b +5 ct2_w
//     +6 fc1_b +7 fc1_w +8 fc2_b +9 fc2_w +10 fc3_b +11 fc3_w
//     +12..15 rb1{b1,b2,w1,w2}
//     +16..23 rb2{b1,b2,bn_b,bn_g,ds_b,ds_w,w1,w2}
//     +24..31 rb3{b1,b2,bn_b,bn_g,ds_b,ds_w,w1,w2}
//     +32 w_in +33 w_out
//   68..89  enc0: 68 b0; 69..76 rb1{b1,b2,bn_b,bn_g,ds_b,ds_w,w1,w2};
//           77..84 rb2{...}; 85..88 rb3{b1,b2,w1,w2}; 89 w0
//   90..95  lt:  b1,b2,b3,w1,w2,w3
//   96..101 rev: b1,b2,b3,w1,w2,w3
//   102     x (1,1,32,65)
// Output: d0(32x64) | d1(32x96) | m0(32x1) | m1(32x2)  = 5216 floats
// ============================================================================

typedef __attribute__((ext_vector_type(16))) _Float16 v16h;
typedef __attribute__((ext_vector_type(8)))  float    v8f;
typedef __attribute__((ext_vector_type(4)))  float    f4;

__device__ __forceinline__ float lrelu(float v) { return v > 0.f ? v : 0.01f * v; }

// ---------------------------------------------------------------------------
// Generic direct conv (NCHW, OIHW), one thread per output element.
// ---------------------------------------------------------------------------
template <int KSZ>
__global__ void k_conv2d(const float* __restrict__ x, const float* __restrict__ w,
                         const float* __restrict__ b, float* __restrict__ out,
                         int Ci, int Co, int Hi, int Wi, int Ho, int Wo,
                         int stride, int pad, int relu)
{
    int idx = blockIdx.x * blockDim.x + threadIdx.x;
    int total = Co * Ho * Wo;
    if (idx >= total) return;
    int ox = idx % Wo; int t = idx / Wo; int oy = t % Ho; int o = t / Ho;
    float acc = b[o];
    const float* wp = w + (size_t)o * Ci * KSZ * KSZ;
    for (int ci = 0; ci < Ci; ++ci) {
        const float* xp = x + (size_t)ci * Hi * Wi;
        const float* wc = wp + ci * KSZ * KSZ;
#pragma unroll
        for (int ky = 0; ky < KSZ; ++ky) {
            int iy = oy * stride + ky - pad;
            if (iy < 0 || iy >= Hi) continue;
#pragma unroll
            for (int kx = 0; kx < KSZ; ++kx) {
                int ix = ox * stride + kx - pad;
                if (ix < 0 || ix >= Wi) continue;
                acc = fmaf(xp[iy * Wi + ix], wc[ky * KSZ + kx], acc);
            }
        }
    }
    out[idx] = relu ? lrelu(acc) : acc;
}

// ---------------------------------------------------------------------------
// Transposed conv 3x3, lhs_dilation=2, pad(1,2): out = 2*in per spatial dim.
// ---------------------------------------------------------------------------
__global__ void k_convt(const float* __restrict__ x, const float* __restrict__ w,
                        const float* __restrict__ b, float* __restrict__ out,
                        int Ci, int Co, int Hi, int Wi, int relu)
{
    int Ho = 2 * Hi, Wo = 2 * Wi;
    int idx = blockIdx.x * blockDim.x + threadIdx.x;
    int total = Co * Ho * Wo;
    if (idx >= total) return;
    int ox = idx % Wo; int t = idx / Wo; int oy = t % Ho; int o = t / Ho;
    float acc = b[o];
#pragma unroll
    for (int ky = 0; ky < 3; ++ky) {
        int a = oy + ky - 1;
        if (a < 0 || a > 2 * Hi - 2 || (a & 1)) continue;
        int iy = a >> 1;
#pragma unroll
        for (int kx = 0; kx < 3; ++kx) {
            int q = ox + kx - 1;
            if (q < 0 || q > 2 * Wi - 2 || (q & 1)) continue;
            int ix = q >> 1;
            int kidx = (2 - ky) * 3 + (2 - kx);
            for (int ci = 0; ci < Ci; ++ci)
                acc = fmaf(x[(size_t)ci * Hi * Wi + iy * Wi + ix],
                           w[((size_t)ci * Co + o) * 9 + kidx], acc);
        }
    }
    out[idx] = relu ? lrelu(acc) : acc;
}

// ---------------------------------------------------------------------------
// Per-channel mean/var (BatchNorm stats). One workgroup per channel.
// ---------------------------------------------------------------------------
__global__ void k_stats(const float* __restrict__ x, float* __restrict__ mean,
                        float* __restrict__ var, int HW)
{
    int c = blockIdx.x;
    const float* p = x + (size_t)c * HW;
    float s = 0.f, ss = 0.f;
    for (int i = threadIdx.x; i < HW; i += blockDim.x) {
        float v = p[i]; s += v; ss += v * v;
    }
    __shared__ float sb[256], qb[256];
    sb[threadIdx.x] = s; qb[threadIdx.x] = ss;
    __syncthreads();
    for (int off = blockDim.x >> 1; off > 0; off >>= 1) {
        if ((int)threadIdx.x < off) { sb[threadIdx.x] += sb[threadIdx.x + off];
                                      qb[threadIdx.x] += qb[threadIdx.x + off]; }
        __syncthreads();
    }
    if (threadIdx.x == 0) {
        float m = sb[0] / HW;
        mean[c] = m;
        var[c] = qb[0] / HW - m * m;
    }
}

__global__ void k_res_lrelu(const float* __restrict__ t2, const float* __restrict__ id,
                            float* __restrict__ out, int n)
{
    int i = blockIdx.x * blockDim.x + threadIdx.x;
    if (i >= n) return;
    out[i] = lrelu(t2[i] + id[i]);
}

__global__ void k_res_bn_lrelu(const float* __restrict__ t2, const float* __restrict__ idc,
                               const float* __restrict__ mean, const float* __restrict__ var,
                               const float* __restrict__ g, const float* __restrict__ bb,
                               float* __restrict__ out, int HW, int total)
{
    int i = blockIdx.x * blockDim.x + threadIdx.x;
    if (i >= total) return;
    int c = i / HW;
    float idn = (idc[i] - mean[c]) * rsqrtf(var[c] + 1e-5f) * g[c] + bb[c];
    out[i] = lrelu(t2[i] + idn);
}

__global__ void k_fill(float* __restrict__ p, int n, float v)
{
    int i = blockIdx.x * blockDim.x + threadIdx.x;
    if (i < n) p[i] = v;
}

// ---------------------------------------------------------------------------
// GEMV (M=1) on v_wmma_f32_16x16x32_f16. 128 threads = 4 waves per block,
// one 16-output tile per wave (wave-uniform -> EXEC all-1s for WMMA).
// K chunked by 64 -> two back-to-back WMMAs per iteration. All fragments
// load as b128; weights use non-temporal loads (read-once stream).
// The loop body is force-inlined twice so the activation pointer's address
// space is static per call site: LDS (ds_load_b128) or global.
// ---------------------------------------------------------------------------
#define SACT_MAX 4096

#if defined(__gfx1250__) && __has_builtin(__builtin_amdgcn_global_load_async_to_lds_b128) && \
    __has_builtin(__builtin_amdgcn_s_wait_asynccnt)
#define HAVE_ASYNC_LDS 1
// Builtin takes pointers to int4 vectors: arg0 AS1 (global), arg1 AS3 (LDS).
typedef int v4i_ls __attribute__((vector_size(16)));
typedef __attribute__((address_space(1))) v4i_ls g_v4i;
typedef __attribute__((address_space(3))) v4i_ls l_v4i;
#else
#define HAVE_ASYNC_LDS 0
#endif

__device__ __forceinline__ v8f
gemv_body(const float* aptr, const float* wrow, int K, int hi)
{
    v8f c = {};
    for (int k0 = 0; k0 < K; k0 += 64) {
        __builtin_prefetch(wrow + k0 + 512, 0, 1);     // global_prefetch_b8
#pragma unroll
        for (int half = 0; half < 2; ++half) {
            const int kc = k0 + half * 32;
            float abuf[16], bbuf[16];
            *(f4*)(abuf + 0)  = *(const f4*)(aptr + kc + hi * 8);
            *(f4*)(abuf + 4)  = *(const f4*)(aptr + kc + hi * 8 + 4);
            *(f4*)(abuf + 8)  = *(const f4*)(aptr + kc + 16 + hi * 8);
            *(f4*)(abuf + 12) = *(const f4*)(aptr + kc + 16 + hi * 8 + 4);
            // read-once weight stream: non-temporal policy
            *(f4*)(bbuf + 0)  = __builtin_nontemporal_load((const f4*)(wrow + kc + hi * 8));
            *(f4*)(bbuf + 4)  = __builtin_nontemporal_load((const f4*)(wrow + kc + hi * 8 + 4));
            *(f4*)(bbuf + 8)  = __builtin_nontemporal_load((const f4*)(wrow + kc + 16 + hi * 8));
            *(f4*)(bbuf + 12) = __builtin_nontemporal_load((const f4*)(wrow + kc + 16 + hi * 8 + 4));
            v16h a, bm;
#pragma unroll
            for (int v = 0; v < 8; ++v) {
                const int ja = (v >> 2) * 8 + (v & 3) * 2;   // A vgpr -> abuf idx
                a[2 * v]      = (_Float16)abuf[ja];
                a[2 * v + 1]  = (_Float16)abuf[ja + 1];
                bm[2 * v]     = (_Float16)bbuf[2 * v];
                bm[2 * v + 1] = (_Float16)bbuf[2 * v + 1];
            }
            c = __builtin_amdgcn_wmma_f32_16x16x32_f16(
                    false, a, false, bm, (short)0, c, false, false);
        }
    }
    return c;
}

__global__ void __launch_bounds__(128)
k_linear(const float* __restrict__ act, const float* __restrict__ w,
         const float* __restrict__ bias, float* __restrict__ out,
         int K, int N, int relu)
{
    __shared__ __align__(16) float sact[SACT_MAX];
    const int tid  = threadIdx.x;
    const int lane = tid & 31;
    const int wave = tid >> 5;
    const int tile = blockIdx.x * (blockDim.x >> 5) + wave;
    const int m    = lane & 15;
    const int hi   = lane >> 4;

    const bool small = (K <= SACT_MAX);
    if (small) {
#if HAVE_ASYNC_LDS
        for (int i = tid * 4; i < K; i += (int)blockDim.x * 4) {
            __builtin_amdgcn_global_load_async_to_lds_b128(
                (g_v4i*)(unsigned long long)(const void*)(act + i),
                (l_v4i*)(unsigned)(unsigned long long)(const void*)(sact + i),
                0, 0);
        }
        __builtin_amdgcn_s_wait_asynccnt(0);
#else
        for (int i = tid; i < K; i += (int)blockDim.x) sact[i] = act[i];
#endif
        __syncthreads();
    }

    if (tile * 16 >= N) return;               // wave-uniform exit
    const int n = tile * 16 + m;
    const float* wrow = w + (size_t)n * K;

    // Two static specializations: address space of the activation pointer is
    // known per call site -> ds_load_b128 (LDS) vs global_load_b128.
    v8f c = small ? gemv_body(sact, wrow, K, hi)
                  : gemv_body(act,  wrow, K, hi);

    if (lane < 16) {                           // D VGPR0, lanes 0..15 => M=0 row
        float r = c[0] + bias[n];
        out[n] = relu ? lrelu(r) : r;
    }
}

// ---------------------------------------------------------------------------
// Final mask/body assembly.
// ---------------------------------------------------------------------------
__global__ void k_finalize(const float* __restrict__ y, float* __restrict__ body_out,
                           float* __restrict__ mask_out, int W, int m)
{
    int bodyW = W - m;                // == 32*(m+1)
    int nbody = 32 * bodyW;
    int idx = blockIdx.x * blockDim.x + threadIdx.x;
    if (idx < nbody) {
        int r = idx / bodyW, col = idx % bodyW;
        int j = col >> 5, cc = col & 31;
        float f = 1.f;
        if (j == 0 && (r < 2 || r >= 30)) f = 0.f;
        if (j < m  && !(y[cc * W + (W - m) + j]     > 0.f)) f = 0.f;
        if (j >= 1 && !(y[r  * W + (W - m) + j - 1] > 0.f)) f = 0.f;
        body_out[idx] = y[r * W + col] * f;
    } else if (idx < nbody + 32 * m) {
        int k = idx - nbody;
        int r = k / m, j = k % m;
        mask_out[k] = (y[r * W + (W - m) + j] > 0.f) ? 1.f : 0.f;
    }
}

// ============================================================================
// Host orchestration
// ============================================================================
static inline int cdiv(int a, int b) { return (a + b - 1) / b; }

extern "C" void kernel_launch(void* const* d_in, const int* in_sizes, int n_in,
                              void* d_out, int out_size, void* d_ws, size_t ws_size,
                              hipStream_t stream)
{
    (void)in_sizes; (void)n_in; (void)out_size; (void)ws_size;
    auto F = [&](int i) { return (const float*)d_in[i]; };
    float* ws  = (float*)d_ws;
    float* W0  = ws;                 // 110592 each: ping-pong activation buffers
    float* W1  = ws + 110592;
    float* W2  = ws + 221184;
    float* W3  = ws + 331776;
    float* CAT = ws + 442368;        // 43008: [e0 | zeros]
    float* R3  = ws + 485376;        // 43008: rev output
    float* MEAN= ws + 528384;        // 256
    float* VAR = ws + 528640;        // 256
    float* V0  = ws + 528896;        // 1024
    float* V1  = ws + 529920;        // 1024
    const int TB = 256;

    auto conv3 = [&](const float* in, int wi, int bi, float* out,
                     int Ci, int Co, int Hi, int Wi, int Ho, int Wo,
                     int s, int relu) {
        int n = Co * Ho * Wo;
        k_conv2d<3><<<cdiv(n, TB), TB, 0, stream>>>(in, F(wi), F(bi), out,
                                                    Ci, Co, Hi, Wi, Ho, Wo, s, 1, relu);
    };
    auto conv1 = [&](const float* in, int wi, int bi, float* out,
                     int Ci, int Co, int Hi, int Wi, int Ho, int Wo,
                     int s, int relu) {
        int n = Co * Ho * Wo;
        k_conv2d<1><<<cdiv(n, TB), TB, 0, stream>>>(in, F(wi), F(bi), out,
                                                    Ci, Co, Hi, Wi, Ho, Wo, s, 0, relu);
    };
    auto lin = [&](const float* a, int wi, int bi, float* out, int K, int N, int relu) {
        k_linear<<<cdiv(N / 16, 4), 128, 0, stream>>>(a, F(wi), F(bi), out, K, N, relu);
    };

    // ---------------- Encoder ----------------
    const float* x = F(102);
    conv3(x, 89, 68, W0, 1, 32, 32, 65, 32, 65, 1, 1);                // h0
    // rb1: 32->64 s2, ds+bn
    conv3(W0, 75, 69, W1, 32, 64, 32, 65, 16, 33, 2, 1);
    conv3(W1, 76, 70, W2, 64, 64, 16, 33, 16, 33, 1, 0);
    conv1(W0, 74, 73, W3, 32, 64, 32, 65, 16, 33, 2, 0);
    k_stats<<<64, 256, 0, stream>>>(W3, MEAN, VAR, 16 * 33);
    { int n = 64 * 16 * 33;
      k_res_bn_lrelu<<<cdiv(n, TB), TB, 0, stream>>>(W2, W3, MEAN, VAR, F(72), F(71), W0, 16 * 33, n); }
    // rb2: 64->128 s2, ds+bn
    conv3(W0, 83, 77, W1, 64, 128, 16, 33, 8, 17, 2, 1);
    conv3(W1, 84, 78, W2, 128, 128, 8, 17, 8, 17, 1, 0);
    conv1(W0, 82, 81, W3, 64, 128, 16, 33, 8, 17, 2, 0);
    k_stats<<<128, 256, 0, stream>>>(W3, MEAN, VAR, 8 * 17);
    { int n = 128 * 8 * 17;
      k_res_bn_lrelu<<<cdiv(n, TB), TB, 0, stream>>>(W2, W3, MEAN, VAR, F(80), F(79), W0, 8 * 17, n); }
    // rb3: 128->128 s1, no ds
    conv3(W0, 87, 85, W1, 128, 128, 8, 17, 8, 17, 1, 1);
    conv3(W1, 88, 86, W2, 128, 128, 8, 17, 8, 17, 1, 0);
    { int n = 128 * 8 * 17;
      k_res_lrelu<<<cdiv(n, TB), TB, 0, stream>>>(W2, W0, CAT, n); }      // e0 -> CAT[0:17408]
    k_fill<<<cdiv(25600, TB), TB, 0, stream>>>(CAT + 17408, 25600, 0.f);  // e1 = zeros

    // ---------------- Latent MLPs (WMMA) ----------------
    lin(CAT, 93, 90, V0, 43008, 512, 1);   // lt.w1
    lin(V0,  94, 91, V1, 512,   256, 1);   // lt.w2
    lin(V1,  95, 92, V0, 256,   128, 1);   // lt.w3
    lin(V0,  99, 96, V1, 128,   256, 1);   // rev.w1
    lin(V1, 100, 97, V0, 256,   512, 1);   // rev.w2
    lin(V0, 101, 98, R3, 512, 43008, 0);   // rev.w3

    // ---------------- Decoders ----------------
    float* out = (float*)d_out;
    auto decoder = [&](int B, const float* s, int Ws, int W, int m,
                       float* bodyOut, float* maskOut) {
        conv3(s, B + 32, B + 0, W0, 128, 64, 2, Ws, 2, Ws, 1, 1);   // w_in
        // rb1 (no ds)
        conv3(W0, B + 14, B + 12, W1, 64, 64, 2, Ws, 2, Ws, 1, 1);
        conv3(W1, B + 15, B + 13, W2, 64, 64, 2, Ws, 2, Ws, 1, 0);
        { int n = 64 * 2 * Ws;
          k_res_lrelu<<<cdiv(n, TB), TB, 0, stream>>>(W2, W0, W1, n); }
        // ct1: (64,2,Ws) -> (64,4,2Ws)
        { int n = 64 * 4 * (2 * Ws);
          k_convt<<<cdiv(n, TB), TB, 0, stream>>>(W1, F(B + 3), F(B + 2), W0, 64, 64, 2, Ws, 1); }
        int H4 = 4, W4 = 2 * Ws;
        // rb2 (ds 64->32, bn)
        conv3(W0, B + 22, B + 16, W1, 64, 32, H4, W4, H4, W4, 1, 1);
        conv3(W1, B + 23, B + 17, W2, 32, 32, H4, W4, H4, W4, 1, 0);
        conv1(W0, B + 21, B + 20, W3, 64, 32, H4, W4, H4, W4, 1, 0);
        k_stats<<<32, 256, 0, stream>>>(W3, MEAN, VAR, H4 * W4);
        { int n = 32 * H4 * W4;
          k_res_bn_lrelu<<<cdiv(n, TB), TB, 0, stream>>>(W2, W3, MEAN, VAR,
                                                         F(B + 19), F(B + 18), W0, H4 * W4, n); }
        // ct2: (32,4,W4) -> (32,8,2W4)
        { int n = 32 * 8 * (2 * W4);
          k_convt<<<cdiv(n, TB), TB, 0, stream>>>(W0, F(B + 5), F(B + 4), W1, 32, 32, H4, W4, 1); }
        int H8 = 8, W8 = 2 * W4;
        // rb3 (ds 32->1, bn)
        conv3(W1, B + 30, B + 24, W2, 32, 1, H8, W8, H8, W8, 1, 1);
        conv3(W2, B + 31, B + 25, W3, 1, 1, H8, W8, H8, W8, 1, 0);
        conv1(W1, B + 29, B + 28, W0, 32, 1, H8, W8, H8, W8, 1, 0);
        k_stats<<<1, 256, 0, stream>>>(W0, MEAN, VAR, H8 * W8);
        { int n = H8 * W8;
          k_res_bn_lrelu<<<cdiv(n, TB), TB, 0, stream>>>(W3, W0, MEAN, VAR,
                                                         F(B + 27), F(B + 26), W2, H8 * W8, n); }
        // w_out
        conv3(W2, B + 33, B + 1, W3, 1, 1, H8, W8, H8, W8, 1, 0);
        // FC head (WMMA)
        int PREV = H8 * W8;            // 2176 / 3200, both %64==0
        lin(W3, B + 7,  B + 6,  V0, PREV, 512, 1);
        lin(V0, B + 9,  B + 8,  V1, 512,  256, 1);
        int CURR = 32 * W;             // 2080 / 3136, both %16==0
        lin(V1, B + 11, B + 10, W0, 256, CURR, 0);
        // mask/body
        int nt = 32 * (W - m) + 32 * m;
        k_finalize<<<cdiv(nt, TB), TB, 0, stream>>>(W0, bodyOut, maskOut, W, m);
    };

    decoder(0,  R3,          68, 65, 1, out + 0,    out + 5120);  // d0, m0
    decoder(34, R3 + 17408, 100, 98, 2, out + 2048, out + 5152);  // d1, m1
}
